// DynamiSE_39316130628234
// MI455X (gfx1250) — matrix operand order repaired and checked by reading, not compiled, not run on hardware
//
#include <hip/hip_runtime.h>
#include <hip/hip_bf16.h>

// ---------------------------------------------------------------------------
// DynamiSE graph-ODE on MI455X (gfx1250, wave32).
//   * GEMMs on v_wmma_f32_16x16x32_f16; N=100000 = 6250 16-row tiles/wave.
//   * All weight B-fragments pre-packed once per launch into per-lane WMMA
//     layout (f16) -> hot kernels load each fragment as one 32B vector load.
//   * LayerNorm fused into A-fragment load via __shfl_xor(.,16) (wave32).
//   * Edge scatter: int32 indices + per-edge coef precomputed once; float4
//     gather + native f32 global atomics (global_atomic_add_f32).
//   * Working set (~130 MB) is L2-resident on the 192 MB L2.
// ---------------------------------------------------------------------------

typedef __attribute__((ext_vector_type(16))) _Float16 v16h;
typedef __attribute__((ext_vector_type(8)))  float    v8f;

#define HID 32

__device__ __forceinline__ v8f wmma_f16(v16h a, v16h b, v8f c) {
  return __builtin_amdgcn_wmma_f32_16x16x32_f16(
      /*neg_a=*/false, a, /*neg_b=*/false, b,
      /*c_mod=*/(short)0, c, /*reuse_a=*/false, /*reuse_b=*/false);
}

// A fragment: 16x32 f16, rows at `base` (f32 source), row stride `ld` floats.
//   slots 0..7  -> K = half*8 + e
//   slots 8..15 -> K = 16 + half*8 + (e-8)
__device__ __forceinline__ v16h loadA(const float* __restrict__ base, int ld, int lane) {
  const int m = lane & 15, half = lane >> 4;
  const float* p = base + (size_t)m * ld;
  v16h a;
#pragma unroll
  for (int e = 0; e < 8; ++e) a[e] = (_Float16)p[half * 8 + e];
#pragma unroll
  for (int e = 0; e < 8; ++e) a[8 + e] = (_Float16)p[16 + half * 8 + e];
  return a;
}

// B fragment from row-major f32 W: col N=lane&15; slots 0..7 -> K = half*8 + e,
// slots 8..15 -> K = 16 + half*8 + (e-8). (Used only in the pack kernel.)
__device__ __forceinline__ v16h loadB(const float* __restrict__ W, int ld, int n0, int lane) {
  const int n = lane & 15, kb = (lane >> 4) * 8;
  v16h b;
#pragma unroll
  for (int e = 0; e < 16; ++e) b[e] = (_Float16)W[(size_t)((e < 8) ? (kb + e) : (16 + kb + e - 8)) * ld + n0 + n];
  return b;
}

__device__ __forceinline__ v16h loadBP(const _Float16* __restrict__ bp, int frag, int lane) {
  return *(const v16h*)(bp + ((size_t)frag * 32 + lane) * 16);
}

__device__ __forceinline__ void atomAddF(float* p, float v) { unsafeAtomicAdd(p, v); }

// --------------------------- precompute kernels ----------------------------

__global__ void zero_f32_kernel(float* __restrict__ p, int n) {
  int i = blockIdx.x * blockDim.x + threadIdx.x;
  if (i < n) p[i] = 0.0f;
}

// int64 [2,E] -> int32 src/dst, and accumulate in-degree (dst counts) as f32.
__global__ void idx_deg_kernel(const long long* __restrict__ ei,
                               int* __restrict__ src, int* __restrict__ dst,
                               float* __restrict__ deg, int E) {
  int e = blockIdx.x * blockDim.x + threadIdx.x;
  if (e >= E) return;
  int s = (int)ei[e];
  int d = (int)ei[E + e];
  src[e] = s;
  dst[e] = d;
  atomAddF(&deg[d], 1.0f);
}

// deg -> dinv = rsqrt(deg+1) (in place), dinv2 = 1/(deg+1)
__global__ void dinv_kernel(float* __restrict__ deg_dinv, float* __restrict__ dinv2, int n) {
  int i = blockIdx.x * blockDim.x + threadIdx.x;
  if (i >= n) return;
  float dg = deg_dinv[i] + 1.0f;
  deg_dinv[i] = rsqrtf(dg);
  dinv2[i] = 1.0f / dg;
}

__global__ void coef_kernel(const int* __restrict__ src, const int* __restrict__ dst,
                            const float* __restrict__ dinv, float* __restrict__ coef, int E) {
  int e = blockIdx.x * blockDim.x + threadIdx.x;
  if (e >= E) return;
  coef[e] = dinv[src[e]] * dinv[dst[e]];
}

// Pack all 12 weight B-fragments into per-lane WMMA layout (f16), and the
// g_ode/b_ode vectors into per-(half,slot) order. One block, 12 waves.
//   frag 0,1 : w_pos  n0=0,16      frag 2,3 : w_neg n0=0,16
//   frag 4,5 : psi rows 0..31      frag 6,7 : psi rows 32..63
//   frag 8,9 : enc rows 0..31      frag10,11: enc rows 32..63
__global__ void pack_kernel(const float* __restrict__ w_pos, const float* __restrict__ w_neg,
                            const float* __restrict__ w_psi, const float* __restrict__ w_enc,
                            const float* __restrict__ g_ode, const float* __restrict__ b_ode,
                            _Float16* __restrict__ bp, float* __restrict__ gbp) {
  int wave = threadIdx.x >> 5;  // 0..11
  int lane = threadIdx.x & 31;
  const float* W;
  switch (wave >> 1) {
    case 0:  W = w_pos; break;
    case 1:  W = w_neg; break;
    case 2:  W = w_psi; break;
    case 3:  W = w_psi + 32 * 32; break;
    case 4:  W = w_enc; break;
    default: W = w_enc + 32 * 32; break;
  }
  int n0 = (wave & 1) * 16;
  v16h b = loadB(W, 32, n0, lane);
  *(v16h*)(bp + ((size_t)wave * 32 + lane) * 16) = b;
  if (wave == 0) {
    int half = lane >> 4, e = lane & 15;
    int k = (e < 8) ? (half * 8 + e) : (16 + half * 8 + (e - 8));
    gbp[lane] = g_ode[k];       // slot-ordered gain
    gbp[32 + lane] = b_ode[k];  // slot-ordered bias
  }
}

// ------------------------------ encoder (h0) -------------------------------

// tmp = x[N,64] @ w_enc[64,32] + b_enc   (per-wave 16-row tile, 4 WMMAs)
__global__ __launch_bounds__(256) void enc_kernel(const float* __restrict__ x,
                                                  const _Float16* __restrict__ bp,
                                                  const float* __restrict__ b_enc,
                                                  float* __restrict__ out, int nTiles) {
  int wave = (blockIdx.x * blockDim.x + threadIdx.x) >> 5;
  int lane = threadIdx.x & 31;
  if (wave >= nTiles) return;
  int row0 = wave * 16;
  int m = lane & 15, half = lane >> 4;

  v16h a0 = loadA(x + (size_t)row0 * 64, 64, lane);       // K = 0..31
  v16h a1 = loadA(x + (size_t)row0 * 64 + 32, 64, lane);  // K = 32..63

#pragma unroll
  for (int t = 0; t < 2; ++t) {
    v16h b0 = loadBP(bp, 8 + t, lane);
    v16h b1 = loadBP(bp, 10 + t, lane);
    v8f c = {};
    c = wmma_f16(a0, b0, c);
    c = wmma_f16(a1, b1, c);
    int col = t * 16 + m;
    float bs = b_enc[col];
#pragma unroll
    for (int r = 0; r < 8; ++r) {
      int row = row0 + 8 * half + r;
      out[(size_t)row * HID + col] = c[r] + bs;
    }
  }
}

// per-node LayerNorm with affine params (used once for h0)
__global__ void ln_kernel(const float* __restrict__ in, const float* __restrict__ g,
                          const float* __restrict__ b, float* __restrict__ out, int n) {
  int i = blockIdx.x * blockDim.x + threadIdx.x;
  if (i >= n) return;
  const float* p = in + (size_t)i * HID;
  float s = 0.f, s2 = 0.f;
#pragma unroll
  for (int k = 0; k < HID; ++k) { float v = p[k]; s += v; s2 += v * v; }
  float mean = s * (1.0f / HID);
  float var = s2 * (1.0f / HID) - mean * mean;
  float rstd = rsqrtf(var + 1e-5f);
  float* o = out + (size_t)i * HID;
#pragma unroll
  for (int k = 0; k < HID; ++k) o[k] = (p[k] - mean) * rstd * g[k] + b[k];
}

// ------------------------------ ode_func stage -----------------------------

__device__ __forceinline__ void gemm_xw_agg(v16h a, const _Float16* __restrict__ bp,
                                            int fragBase, const float* __restrict__ bias,
                                            const float* __restrict__ dinv2,
                                            float* __restrict__ xw, float* __restrict__ agg,
                                            int row0, int lane) {
  int m = lane & 15, half = lane >> 4;
  float dv[8];
#pragma unroll
  for (int r = 0; r < 8; ++r) dv[r] = dinv2[row0 + 8 * half + r];
#pragma unroll
  for (int t = 0; t < 2; ++t) {
    v16h b = loadBP(bp, fragBase + t, lane);
    v8f c = {};
    c = wmma_f16(a, b, c);
    int col = t * 16 + m;
    float bs = bias[col];
#pragma unroll
    for (int r = 0; r < 8; ++r) {
      int row = row0 + 8 * half + r;
      size_t idx = (size_t)row * HID + col;
      float xv = c[r];
      xw[idx] = xv;
      agg[idx] = xv * dv[r] + bs;  // self-loop term + conv bias
    }
  }
}

// hn = LN(hin)*g+b fused into A-fragment; xw_{p,n} = hn @ W via WMMA;
// agg_{p,n} initialized with self-term + bias.
__global__ __launch_bounds__(256) void pre_kernel(
    const float* __restrict__ hin, float* __restrict__ hn,
    float* __restrict__ xwP, float* __restrict__ xwN,
    float* __restrict__ aggP, float* __restrict__ aggN,
    const _Float16* __restrict__ bp, const float* __restrict__ gbp,
    const float* __restrict__ b_pos, const float* __restrict__ b_neg,
    const float* __restrict__ dinv2P, const float* __restrict__ dinv2N, int nTiles) {
  int wave = (blockIdx.x * blockDim.x + threadIdx.x) >> 5;
  int lane = threadIdx.x & 31;
  if (wave >= nTiles) return;
  int row0 = wave * 16;
  int m = lane & 15, half = lane >> 4;

  // this lane covers two 8-runs of row (row0+m); lane^16 covers the rest ->
  // one xor-16 shuffle completes the 32-element row statistics.
  const float* p = hin + (size_t)(row0 + m) * HID;
  float v[16];
#pragma unroll
  for (int e = 0; e < 8; ++e) v[e] = p[half * 8 + e];
#pragma unroll
  for (int e = 0; e < 8; ++e) v[8 + e] = p[16 + half * 8 + e];
  float s = 0.f, s2 = 0.f;
#pragma unroll
  for (int e = 0; e < 16; ++e) { s += v[e]; s2 += v[e] * v[e]; }
  s  += __shfl_xor(s, 16, 32);
  s2 += __shfl_xor(s2, 16, 32);
  float mean = s * (1.0f / HID);
  float var = s2 * (1.0f / HID) - mean * mean;
  float rstd = rsqrtf(var + 1e-5f);

  const float* gsl = gbp + half * 16;       // slot-ordered g_ode (16 consecutive)
  const float* bsl = gbp + 32 + half * 16;  // slot-ordered b_ode
  v16h a;
  float* hnrow = hn + (size_t)(row0 + m) * HID;
  float xn[16];
#pragma unroll
  for (int e = 0; e < 16; ++e) {
    float xv = (v[e] - mean) * rstd * gsl[e] + bsl[e];
    xn[e] = xv;
    a[e] = (_Float16)xv;
  }
  // write hn back: slots map to two contiguous 8-float runs
#pragma unroll
  for (int e = 0; e < 8; ++e) hnrow[half * 8 + e] = xn[e];
#pragma unroll
  for (int e = 0; e < 8; ++e) hnrow[16 + half * 8 + e] = xn[8 + e];

  gemm_xw_agg(a, bp, 0, b_pos, dinv2P, xwP, aggP, row0, lane);
  gemm_xw_agg(a, bp, 2, b_neg, dinv2N, xwN, aggN, row0, lane);
}

// 8 threads per edge, 4 channels each: float4 gather + 4 native f32 atomics.
__global__ __launch_bounds__(256) void scatter_kernel(
    const int* __restrict__ src, const int* __restrict__ dst,
    const float* __restrict__ coef, const float* __restrict__ xw,
    float* __restrict__ agg, int E) {
  int tid = blockIdx.x * blockDim.x + threadIdx.x;
  int e = tid >> 3;
  if (e >= E) return;
  int c0 = (tid & 7) * 4;
  int s = src[e], d = dst[e];
  float cf = coef[e];
  const float4 v = *(const float4*)(xw + (size_t)s * HID + c0);
  float* o = agg + (size_t)d * HID + c0;
  atomAddF(o + 0, cf * v.x);
  atomAddF(o + 1, cf * v.y);
  atomAddF(o + 2, cf * v.z);
  atomAddF(o + 3, cf * v.w);
}

// delta = tanh([hp|hm] @ w_psi); k = clip(delta - 0.1*hn, -50, 50); fused RK4:
//   stage 0: acc = k;        htmp = h + 0.5*dt*k
//   stage 1: acc += 2k;      htmp = h + 0.5*dt*k
//   stage 2: acc += 2k;      htmp = h + dt*k
//   stage 3: h  += dt/6 * (acc + k)
__global__ __launch_bounds__(256) void post_kernel(
    const float* __restrict__ hp, const float* __restrict__ hm,
    const _Float16* __restrict__ bp, const float* __restrict__ hn,
    float* __restrict__ h, float* __restrict__ htmp, float* __restrict__ acc,
    const float* __restrict__ t, int stage, int nTiles) {
  int wave = (blockIdx.x * blockDim.x + threadIdx.x) >> 5;
  int lane = threadIdx.x & 31;
  if (wave >= nTiles) return;
  int row0 = wave * 16;
  int m = lane & 15, half = lane >> 4;

  v16h ahp = loadA(hp + (size_t)row0 * HID, HID, lane);
  v16h ahm = loadA(hm + (size_t)row0 * HID, HID, lane);

  float dt = (t[1] - t[0]) * 0.125f;  // (t1-t0)/RK4_STEPS
  float cstep = (stage == 2) ? dt : 0.5f * dt;

#pragma unroll
  for (int tt = 0; tt < 2; ++tt) {
    v16h btop = loadBP(bp, 4 + tt, lane);  // psi rows 0..31  (hp part)
    v16h bbot = loadBP(bp, 6 + tt, lane);  // psi rows 32..63 (hm part)
    v8f c = {};
    c = wmma_f16(ahp, btop, c);
    c = wmma_f16(ahm, bbot, c);
    int col = tt * 16 + m;
#pragma unroll
    for (int r = 0; r < 8; ++r) {
      int row = row0 + 8 * half + r;
      size_t idx = (size_t)row * HID + col;
      float k = tanhf(c[r]) - 0.1f * hn[idx];
      k = fminf(fmaxf(k, -50.0f), 50.0f);
      if (stage == 0) {
        acc[idx] = k;
        htmp[idx] = h[idx] + cstep * k;
      } else if (stage < 3) {
        acc[idx] += 2.0f * k;
        htmp[idx] = h[idx] + cstep * k;
      } else {
        h[idx] += (dt * (1.0f / 6.0f)) * (acc[idx] + k);
      }
    }
  }
}

// ------------------------------- launcher ----------------------------------

extern "C" void kernel_launch(void* const* d_in, const int* in_sizes, int n_in,
                              void* d_out, int out_size, void* d_ws, size_t ws_size,
                              hipStream_t stream) {
  const float*     x     = (const float*)d_in[0];
  const long long* eiP   = (const long long*)d_in[1];
  const long long* eiN   = (const long long*)d_in[2];
  const float*     t     = (const float*)d_in[3];
  const float*     w_enc = (const float*)d_in[4];
  const float*     b_enc = (const float*)d_in[5];
  const float*     g_feat= (const float*)d_in[6];
  const float*     b_feat= (const float*)d_in[7];
  const float*     w_pos = (const float*)d_in[8];
  const float*     b_pos = (const float*)d_in[9];
  const float*     w_neg = (const float*)d_in[10];
  const float*     b_neg = (const float*)d_in[11];
  const float*     w_psi = (const float*)d_in[12];
  const float*     g_ode = (const float*)d_in[13];
  const float*     b_ode = (const float*)d_in[14];
  float* h = (float*)d_out;

  const int N = in_sizes[0] / 64;  // 100000
  const int E = in_sizes[1] / 2;   // 1600000

  // carve workspace (256B aligned chunks)
  char* ws = (char*)d_ws;
  size_t off = 0;
  auto carve = [&](size_t bytes) -> char* {
    char* p = ws + off;
    off += (bytes + 255) & ~(size_t)255;
    return p;
  };
  int*   srcP  = (int*)carve((size_t)E * 4);
  int*   dstP  = (int*)carve((size_t)E * 4);
  int*   srcN  = (int*)carve((size_t)E * 4);
  int*   dstN  = (int*)carve((size_t)E * 4);
  float* coefP = (float*)carve((size_t)E * 4);
  float* coefN = (float*)carve((size_t)E * 4);
  float* dinvP = (float*)carve((size_t)N * 4);  // holds deg first, then rsqrt
  float* dinvN = (float*)carve((size_t)N * 4);
  float* din2P = (float*)carve((size_t)N * 4);
  float* din2N = (float*)carve((size_t)N * 4);
  size_t NH = (size_t)N * HID * 4;
  float* hn   = (float*)carve(NH);
  float* xwP  = (float*)carve(NH);
  float* xwN  = (float*)carve(NH);
  float* aggP = (float*)carve(NH);
  float* aggN = (float*)carve(NH);
  float* htmp = (float*)carve(NH);
  float* acc  = (float*)carve(NH);
  _Float16* bpack = (_Float16*)carve((size_t)12 * 32 * 16 * 2);  // 12 frags
  float*    gbp   = (float*)carve((size_t)64 * 4);               // g/b slot order
  (void)ws_size; (void)n_in; (void)out_size;

  const int THR = 256;
  dim3 blk(THR);
  dim3 gN((N + THR - 1) / THR);
  dim3 gE((E + THR - 1) / THR);
  dim3 gE8(((size_t)E * 8 + THR - 1) / THR);
  const int nTiles = N / 16;                    // 100000/16 = 6250 (exact)
  dim3 gW((nTiles * 32 + THR - 1) / THR);       // one wave per 16-row tile

  // ---- loop-invariant precompute: indices, degrees, coefs, packed weights --
  zero_f32_kernel<<<gN, blk, 0, stream>>>(dinvP, N);
  zero_f32_kernel<<<gN, blk, 0, stream>>>(dinvN, N);
  idx_deg_kernel<<<gE, blk, 0, stream>>>(eiP, srcP, dstP, dinvP, E);
  idx_deg_kernel<<<gE, blk, 0, stream>>>(eiN, srcN, dstN, dinvN, E);
  dinv_kernel<<<gN, blk, 0, stream>>>(dinvP, din2P, N);
  dinv_kernel<<<gN, blk, 0, stream>>>(dinvN, din2N, N);
  coef_kernel<<<gE, blk, 0, stream>>>(srcP, dstP, dinvP, coefP, E);
  coef_kernel<<<gE, blk, 0, stream>>>(srcN, dstN, dinvN, coefN, E);
  pack_kernel<<<1, 384, 0, stream>>>(w_pos, w_neg, w_psi, w_enc, g_ode, b_ode,
                                     bpack, gbp);

  // ---- h0 = LN(x @ w_enc + b_enc) ----
  enc_kernel<<<gW, blk, 0, stream>>>(x, bpack, b_enc, aggP /*tmp*/, nTiles);
  ln_kernel<<<gN, blk, 0, stream>>>(aggP, g_feat, b_feat, h, N);

  // ---- 8 RK4 steps x 4 stages ----
  for (int step = 0; step < 8; ++step) {
    for (int stage = 0; stage < 4; ++stage) {
      const float* hin = (stage == 0) ? h : htmp;
      pre_kernel<<<gW, blk, 0, stream>>>(hin, hn, xwP, xwN, aggP, aggN,
                                         bpack, gbp, b_pos, b_neg,
                                         din2P, din2N, nTiles);
      scatter_kernel<<<gE8, blk, 0, stream>>>(srcP, dstP, coefP, xwP, aggP, E);
      scatter_kernel<<<gE8, blk, 0, stream>>>(srcN, dstN, coefN, xwN, aggN, E);
      post_kernel<<<gW, blk, 0, stream>>>(aggP, aggN, bpack, hn, h, htmp, acc,
                                          t, stage, nTiles);
    }
  }
}
